// LSTMHardCell_51969104281664
// MI455X (gfx1250) — matrix operand order repaired and checked by reading, not compiled, not run on hardware
//
#include <hip/hip_runtime.h>
#include <hip/hip_bf16.h>
#include <stdint.h>

typedef __bf16 bf16_t;
typedef bf16_t v4bf  __attribute__((ext_vector_type(4)));
typedef bf16_t v8bf  __attribute__((ext_vector_type(8)));
typedef bf16_t v16bf __attribute__((ext_vector_type(16)));
typedef float  v4f   __attribute__((ext_vector_type(4)));
typedef float  v8f   __attribute__((ext_vector_type(8)));
typedef int    v4i   __attribute__((ext_vector_type(4)));

union Frag { v16bf v; v8bf h[2]; };

#define B_DIM 4096
#define H_DIM 2048
#define K_DIM 4096   /* I + H, single fused-K GEMM */
#define BM 256       /* block rows (batch) */
#define BN 32        /* block cols (hidden) */
#define BK 32        /* K step */
#define LDA 40       /* bf16 elems: 32 + 8 pad -> 80B rows, 16B aligned */
#define LDB 40

#if __has_builtin(__builtin_amdgcn_global_load_async_to_lds_b128)
#define ASYNC_LDS 1
#endif

__device__ __forceinline__ void copy16_to_lds(bf16_t* dst, const bf16_t* src) {
#ifdef ASYNC_LDS
  __builtin_amdgcn_global_load_async_to_lds_b128(
      (__attribute__((address_space(1))) v4i*)src,
      (__attribute__((address_space(3))) v4i*)dst, 0, 0);
#else
  *(v8bf*)dst = *(const v8bf*)src;
#endif
}

__device__ __forceinline__ void wait_stage() {
#ifdef ASYNC_LDS
#if __has_builtin(__builtin_amdgcn_s_wait_asynccnt)
  __builtin_amdgcn_s_wait_asynccnt(0);
#else
  asm volatile("s_wait_asynccnt 0" ::: "memory");
#endif
#endif
  __syncthreads();
}

// ---------------- pack kernels: fp32 -> bf16, K-concatenated ----------------

__global__ __launch_bounds__(256) void pack_A_kernel(
    const float* __restrict__ x, const float* __restrict__ state,
    bf16_t* __restrict__ A16) {
  uint64_t e = ((uint64_t)blockIdx.x * 256u + threadIdx.x) * 4u;
  if (e >= (uint64_t)B_DIM * K_DIM) return;
  uint32_t b = (uint32_t)(e >> 12);          // K_DIM = 2^12
  uint32_t k = (uint32_t)(e & (K_DIM - 1));  // multiple of 4, never crosses 2048
  const float* src = (k < H_DIM) ? (x + (uint64_t)b * H_DIM + k)
                                 : (state + (uint64_t)b * H_DIM + (k - H_DIM));
  v4f f = *(const v4f*)src;
  v4bf o;
  o[0] = (bf16_t)f[0]; o[1] = (bf16_t)f[1]; o[2] = (bf16_t)f[2]; o[3] = (bf16_t)f[3];
  *(v4bf*)(A16 + e) = o;
}

__global__ __launch_bounds__(256) void pack_W_kernel(
    const float* __restrict__ W_in, const float* __restrict__ W_h,
    bf16_t* __restrict__ W16) {
  uint64_t e = ((uint64_t)blockIdx.x * 256u + threadIdx.x) * 4u;
  if (e >= (uint64_t)4 * H_DIM * K_DIM) return;
  uint32_t g   = (uint32_t)(e >> 23);              // per-gate 2048*4096 = 2^23
  uint64_t rem = e & ((1ull << 23) - 1);
  uint32_t h = (uint32_t)(rem >> 12);
  uint32_t k = (uint32_t)(rem & (K_DIM - 1));
  const float* src = (k < H_DIM)
      ? (W_in + ((uint64_t)g << 22) + (uint64_t)h * H_DIM + k)
      : (W_h  + ((uint64_t)g << 22) + (uint64_t)h * H_DIM + (k - H_DIM));
  v4f f = *(const v4f*)src;
  v4bf o;
  o[0] = (bf16_t)f[0]; o[1] = (bf16_t)f[1]; o[2] = (bf16_t)f[2]; o[3] = (bf16_t)f[3];
  *(v4bf*)(W16 + e) = o;
}

// --------------- fused GEMM (4 gates) + LSTM-hard epilogue ------------------

__global__ __launch_bounds__(256, 1) void lstm_gemm_fused(
    const bf16_t* __restrict__ A16, const bf16_t* __restrict__ W16,
    const float* __restrict__ cell,
    float* __restrict__ out_state, float* __restrict__ out_cell) {
  __shared__ __attribute__((aligned(16))) bf16_t Asm[2][BM][LDA];   // 40 KB
  __shared__ __attribute__((aligned(16))) bf16_t Bsm[2][4][BN][LDB];// 20 KB

  const uint32_t tid  = threadIdx.x;
  const uint32_t wave = tid >> 5;
  const uint32_t lane = tid & 31;
  const uint32_t khi  = lane >> 4;   // which half of the wave
  const uint32_t r16  = lane & 15;

  const uint32_t blockH = blockIdx.x * BN;
  const uint32_t blockB = blockIdx.y * BM;

  auto stageLoad = [&](uint32_t s, uint32_t k0) {
    // A tile: BM x BK bf16 = 1024 x 16B chunks; 4 per thread
#pragma unroll
    for (int r = 0; r < 4; ++r) {
      uint32_t c = tid + r * 256u;
      uint32_t row = c >> 2, cc = c & 3;
      copy16_to_lds(&Asm[s][row][cc * 8],
                    A16 + (uint64_t)(blockB + row) * K_DIM + k0 + cc * 8);
    }
    // B tiles: 4 gates x BN x BK bf16 = 512 x 16B chunks; 2 per thread
#pragma unroll
    for (int r = 0; r < 2; ++r) {
      uint32_t c = tid + r * 256u;
      uint32_t g = c >> 7, row = (c >> 2) & 31, cc = c & 3;
      copy16_to_lds(&Bsm[s][g][row][cc * 8],
                    W16 + ((uint64_t)g * H_DIM + blockH + row) * K_DIM + k0 + cc * 8);
    }
  };

  // acc[mt][ct][g] : wave tile = 32 rows x 32 cols x 4 gates
  v8f acc[2][2][4];
#pragma unroll
  for (int mt = 0; mt < 2; ++mt)
#pragma unroll
    for (int ct = 0; ct < 2; ++ct)
#pragma unroll
      for (int g = 0; g < 4; ++g)
        acc[mt][ct][g] = v8f{0.f,0.f,0.f,0.f,0.f,0.f,0.f,0.f};

  auto compute = [&](uint32_t cur) {
    // A fragments (16x32 bf16): row = r16, K chunks at khi*8 and khi*8+16
    Frag fa[2];
#pragma unroll
    for (int mt = 0; mt < 2; ++mt) {
      const bf16_t* abase = &Asm[cur][wave * 32 + mt * 16 + r16][khi * 8];
      fa[mt].h[0] = *(const v8bf*)(abase);
      fa[mt].h[1] = *(const v8bf*)(abase + 16);
    }
#pragma unroll
    for (int ct = 0; ct < 2; ++ct) {
#pragma unroll
      for (int g = 0; g < 4; ++g) {
        // B fragment (32x16 bf16): col = r16, K chunks at khi*8 and khi*8+16
        Frag fb;
        const bf16_t* bbase = &Bsm[cur][g][ct * 16 + r16][khi * 8];
        fb.h[0] = *(const v8bf*)(bbase);
        fb.h[1] = *(const v8bf*)(bbase + 16);
#pragma unroll
        for (int mt = 0; mt < 2; ++mt) {
          acc[mt][ct][g] = __builtin_amdgcn_wmma_f32_16x16x32_bf16(
              false, fa[mt].v, false, fb.v, (short)0, acc[mt][ct][g], false, false);
        }
      }
    }
  };

  const uint32_t NK = K_DIM / BK;  // 128 K-steps
  stageLoad(0, 0);
  for (uint32_t kt = 0; kt < NK - 1; ++kt) {     // steady state: always prefetch
    wait_stage();                                // asyncs of `cur` done + barrier
    stageLoad((kt + 1) & 1, (kt + 1) * BK);
    compute(kt & 1);
  }
  wait_stage();
  compute((NK - 1) & 1);                         // peeled last step, no branch in loop

  // Epilogue (C/D layout: M = v + 8*khi, N = r16)
#pragma unroll
  for (int mt = 0; mt < 2; ++mt) {
#pragma unroll
    for (int ct = 0; ct < 2; ++ct) {
      const uint32_t h = blockH + ct * 16 + r16;
#pragma unroll
      for (int v = 0; v < 8; ++v) {
        const uint32_t b = blockB + wave * 32 + mt * 16 + v + khi * 8;
        const uint64_t off = (uint64_t)b * H_DIM + h;
        const float gf = acc[mt][ct][0][v], gi = acc[mt][ct][1][v];
        const float go = acc[mt][ct][2][v], gu = acc[mt][ct][3][v];
        const float f = 1.0f / (1.0f + __expf(-gf));
        const float i = 1.0f / (1.0f + __expf(-gi));
        const float o = 1.0f / (1.0f + __expf(-go));
        const float u = fminf(fmaxf(gu, -1.0f), 1.0f);
        const float c_old = cell[off];
        const float nc = f * c_old + i * u;
        const float ns = o * fminf(fmaxf(nc, -1.0f), 1.0f);
        out_state[off] = ns;
        out_cell[off]  = nc;
      }
    }
  }
}

extern "C" void kernel_launch(void* const* d_in, const int* in_sizes, int n_in,
                              void* d_out, int out_size, void* d_ws, size_t ws_size,
                              hipStream_t stream) {
  const float* x     = (const float*)d_in[0];
  const float* state = (const float*)d_in[1];
  const float* cell  = (const float*)d_in[2];
  const float* W_in  = (const float*)d_in[3];
  const float* W_h   = (const float*)d_in[4];

  float* out_state = (float*)d_out;
  float* out_cell  = out_state + (size_t)B_DIM * H_DIM;

  bf16_t* A16 = (bf16_t*)d_ws;                       // [4096][4096] bf16
  bf16_t* W16 = A16 + (size_t)B_DIM * K_DIM;         // [4][2048][4096] bf16

  pack_A_kernel<<<(B_DIM * (size_t)K_DIM / 4 + 255) / 256, 256, 0, stream>>>(x, state, A16);
  pack_W_kernel<<<(4ull * H_DIM * K_DIM / 4 + 255) / 256, 256, 0, stream>>>(W_in, W_h, W16);

  dim3 grid(H_DIM / BN, B_DIM / BM);                 // (64, 16)
  lstm_gemm_fused<<<grid, 256, 0, stream>>>(A16, W16, cell, out_state, out_cell);
}
